// DenseKAN_4647154614943
// MI455X (gfx1250) — matrix operand order repaired and checked by reading, not compiled, not run on hardware
//
#include <hip/hip_runtime.h>

// DenseKAN: out[b,o] = sum_{i,k} Bspline[b,i,k] * W[i,k,o]
//   == GEMM  M=4096, N=1024, K=8192  with A generated from x.
// Phase 1 builds A (f16) and W^T (f16) in workspace; phase 2 is a WMMA-f16
// tiled GEMM (f32 accum) with double-buffered LDS fed by CDNA5
// global_load_async_to_lds_b128 (ASYNCcnt-tracked, no VGPR staging).

#define BATCH   4096
#define INSZ    1024
#define UNITS   1024
#define NBAS    8               // GRID_SIZE + SPLINE_ORDER
#define KDIM    (INSZ * NBAS)   // 8192

typedef __attribute__((ext_vector_type(16))) _Float16 v16h;
typedef __attribute__((ext_vector_type(8)))  float    v8f;

// ---------------------------------------------------------------------------
// Phase 1a: cubic B-spline basis values -> A (f16), row-major [BATCH][KDIM].
// Uniform knots g[j] = -2.2 + 0.4*j (j=0..11), Cox–de Boor fully unrolled.
// ---------------------------------------------------------------------------
__global__ __launch_bounds__(256) void kan_basis(const float* __restrict__ x,
                                                 _Float16* __restrict__ A)
{
    const int idx = blockIdx.x * 256 + threadIdx.x;   // over BATCH*INSZ
    const float xv = x[idx];

    float b[11];
#pragma unroll
    for (int j = 0; j < 11; ++j) {
        const float g0 = -2.2f + 0.4f * j;
        const float g1 = g0 + 0.4f;
        b[j] = (xv >= g0 && xv < g1) ? 1.0f : 0.0f;
    }
#pragma unroll
    for (int k = 1; k <= 3; ++k) {
        const float inv = 1.0f / (0.4f * (float)k);
#pragma unroll
        for (int j = 0; j + k < 11; ++j) {            // ascending in-place is safe
            const float gj   = -2.2f + 0.4f * j;
            const float gjk1 = -2.2f + 0.4f * (float)(j + k + 1);
            const float left  = (xv - gj)   * inv;
            const float right = (gjk1 - xv) * inv;
            b[j] = left * b[j] + right * b[j + 1];
        }
    }

    union { _Float16 h[8]; uint4 u; } out;
#pragma unroll
    for (int j = 0; j < 8; ++j) out.h[j] = (_Float16)b[j];
    *(uint4*)(A + (size_t)idx * 8) = out.u;           // 16B coalesced store
}

// ---------------------------------------------------------------------------
// Phase 1b: W (i,k,o) f32  ->  Bt[o][i*8+k] f16  (row-major [UNITS][KDIM]).
// Gives the GEMM identical K-contiguous layouts for both operands.
// ---------------------------------------------------------------------------
__global__ __launch_bounds__(256) void kan_wt(const float* __restrict__ W,
                                              _Float16* __restrict__ Bt)
{
    const int o = blockIdx.x * 256 + threadIdx.x;     // gridDim.x = UNITS/256
    const int i = blockIdx.y;                          // gridDim.y = INSZ

    union { _Float16 h[8]; uint4 u; } out;
#pragma unroll
    for (int k = 0; k < 8; ++k)
        out.h[k] = (_Float16)W[(size_t)i * (8 * UNITS) + (size_t)k * UNITS + o];
    *(uint4*)(Bt + (size_t)o * KDIM + i * 8) = out.u;
}

// ---------------------------------------------------------------------------
// Phase 2: C[m][n] = sum_k A[m][k] * Bt[n][k]   (f16 in, f32 out)
// 256 threads = 8 wave32s (2Mx4N); block tile 128x128, K-step 32; each wave:
// 64x32 via 4x2 v_wmma_f32_16x16x32_f16. Double-buffered LDS filled with
// global_load_async_to_lds_b128; one barrier + s_wait_asynccnt per K-step.
// ---------------------------------------------------------------------------
#define BM   128
#define BN   128
#define BK   32
#define LDT  80        // bytes per LDS tile row: 32 halfs (64B) + 16B pad
#define ATILE (BM * LDT)        // 10240 B
#define BUFB  (2 * ATILE)       // 20480 B per double-buffer stage (A + B tile)

__device__ __forceinline__ void async_ld128(unsigned lds_off, const void* g)
{
    asm volatile("global_load_async_to_lds_b128 %0, %1, off"
                 :: "v"(lds_off), "v"((unsigned long long)(size_t)g)
                 : "memory");
}
__device__ __forceinline__ void wait_async0()
{
    asm volatile("s_wait_asynccnt 0x0" ::: "memory");
}

__global__ __launch_bounds__(256) void kan_gemm(const _Float16* __restrict__ A,
                                                const _Float16* __restrict__ Bt,
                                                float* __restrict__ C)
{
    __shared__ __align__(16) unsigned char smem[2 * BUFB];   // 40 KiB

    const int t      = threadIdx.x;
    const int wave   = t >> 5;
    const int lane   = t & 31;
    const int waveM  = wave >> 2;          // 0..1
    const int waveN  = wave & 3;           // 0..3
    const int laneLo = lane & 15;
    const int laneHi = lane >> 4;

    const int m0 = blockIdx.y * BM;
    const int n0 = blockIdx.x * BN;

    // staging assignment: 2 threads per tile row, 32 bytes each
    const int lr = t >> 1;                 // row 0..127
    const int lh = t & 1;                  // which 32B half of the 64B row
    const _Float16* gA = A  + (size_t)(m0 + lr) * KDIM + lh * 16;
    const _Float16* gB = Bt + (size_t)(n0 + lr) * KDIM + lh * 16;

    // generic shared pointers keep the LDS byte offset in the low dword
    const unsigned sAoff = (unsigned)(size_t)smem + lr * LDT + lh * 32;
    const unsigned sBoff = sAoff + ATILE;

    v8f acc[4][2] = {};

    // stage tile at k0 into buffer buf (4 async b128 per thread = 16KB/block)
    auto stage = [&](int k0, int buf) {
        const unsigned o = (unsigned)(buf * BUFB);
        async_ld128(sAoff + o,      gA + k0);
        async_ld128(sAoff + o + 16, gA + k0 + 8);
        async_ld128(sBoff + o,      gB + k0);
        async_ld128(sBoff + o + 16, gB + k0 + 8);
    };

    stage(0, 0);
    wait_async0();
    __syncthreads();

    int cur = 0;
    for (int k0 = 0; k0 < KDIM; k0 += BK) {
        if (k0 + BK < KDIM) stage(k0 + BK, cur ^ 1);   // fill other buffer
        if (k0 + 2 * BK < KDIM) {                      // warm L2 two tiles out
            __builtin_prefetch((const void*)(gA + k0 + 2 * BK), 0, 1);
            __builtin_prefetch((const void*)(gB + k0 + 2 * BK), 0, 1);
        }

        const unsigned char* base = smem + cur * BUFB;

        // A-fragment (16x32 f16): lanes 0-15 hold K=0-7 & 16-23,
        //                         lanes 16-31 hold K=8-15 & 24-31.
        v16h af[4];
#pragma unroll
        for (int am = 0; am < 4; ++am) {
            const unsigned char* p =
                base + (waveM * 64 + am * 16 + laneLo) * LDT + laneHi * 16;
            ((float4*)&af[am])[0] = *(const float4*)(p);
            ((float4*)&af[am])[1] = *(const float4*)(p + 32);
        }
        // B-fragment (32x16 f16): lanes 0-15 hold K=0-7 & 16-23,
        //                         lanes 16-31 hold K=8-15 & 24-31.
        v16h bf[2];
#pragma unroll
        for (int bn = 0; bn < 2; ++bn) {
            const unsigned char* p =
                base + ATILE + (waveN * 32 + bn * 16 + laneLo) * LDT + laneHi * 16;
            ((float4*)&bf[bn])[0] = *(const float4*)(p);
            ((float4*)&bf[bn])[1] = *(const float4*)(p + 32);
        }

#pragma unroll
        for (int am = 0; am < 4; ++am)
#pragma unroll
            for (int bn = 0; bn < 2; ++bn)
                acc[am][bn] = __builtin_amdgcn_wmma_f32_16x16x32_f16(
                    false, af[am], false, bf[bn],
                    (short)0, acc[am][bn], false, false);

        wait_async0();        // next tile fully landed in LDS (this wave)
        __syncthreads();      // ... and in every wave
        cur ^= 1;
    }

    // C/D layout: VGPR j -> M = j + 8*laneHi, N = laneLo ; streaming NT stores
#pragma unroll
    for (int am = 0; am < 4; ++am)
#pragma unroll
        for (int bn = 0; bn < 2; ++bn)
#pragma unroll
            for (int j = 0; j < 8; ++j) {
                const int m = m0 + waveM * 64 + am * 16 + laneHi * 8 + j;
                const int n = n0 + waveN * 32 + bn * 16 + laneLo;
                __builtin_nontemporal_store(acc[am][bn][j],
                                            &C[(size_t)m * UNITS + n]);
            }
}

// ---------------------------------------------------------------------------
extern "C" void kernel_launch(void* const* d_in, const int* in_sizes, int n_in,
                              void* d_out, int out_size, void* d_ws, size_t ws_size,
                              hipStream_t stream) {
    const float* x = (const float*)d_in[0];                 // [4096,1024] f32
    const float* W = (const float*)d_in[1];                 // [1024,8,1024] f32
    float* out = (float*)d_out;                             // [4096,1024] f32

    // workspace: A f16 (64 MiB) then Bt f16 (16 MiB)
    _Float16* Afp16 = (_Float16*)d_ws;
    _Float16* Btf16 = (_Float16*)((char*)d_ws + (size_t)BATCH * KDIM * sizeof(_Float16));

    kan_basis<<<(BATCH * INSZ) / 256, 256, 0, stream>>>(x, Afp16);
    kan_wt<<<dim3(UNITS / 256, INSZ), 256, 0, stream>>>(W, Btf16);
    kan_gemm<<<dim3(UNITS / BN, BATCH / BM), 256, 0, stream>>>(Afp16, Btf16, out);
}
